// EncoderDCRNN_7490422964884
// MI455X (gfx1250) — compile-verified, matrix operand order checked by reading, not run on hardware
//
#include <hip/hip_runtime.h>
#include <stdint.h>
#include <math.h>

// ---------------------------------------------------------------------------
// DCRNN encoder for MI455X (gfx1250, wave32, WMMA, TDM).
//  - fp32 atomic scatter-add SpMM for K-hop diffusion (L2-resident state)
//  - f16 v_wmma_f32_16x16x32_f16 GEMMs; K folded into one padded A operand
//  - B^T weights ([64 x SK+8] f16, bank-conflict-free padded stride) loaded
//    whole into LDS once per block via TENSOR_LOAD_TO_LDS (TDM) +
//    s_wait_tensorcnt; K-loop is barrier-free, fully unrolled, with B
//    fragments register-staged so WMMAs issue back-to-back
//  - fused GRU epilogues (sigmoid/tanh, r*h write-back, in-place h update)
// ---------------------------------------------------------------------------

typedef __attribute__((ext_vector_type(16))) _Float16 v16h;
typedef __attribute__((ext_vector_type(8)))  float    v8f;
typedef __attribute__((ext_vector_type(4)))  float    f32x4;
typedef __attribute__((ext_vector_type(4)))  unsigned u32x4;
typedef __attribute__((ext_vector_type(8)))  int      i32x8;
typedef __attribute__((ext_vector_type(4)))  int      i32x4;

#define NB 8
#define NN 10000
#define NT_STEPS 12
#define NF 2
#define NH 64
#define NE 160000
#define BN (NB*NN)        // 80000 rows
#define FIN0 (NF+NH)      // 66
#define FIN1 (NH+NH)      // 128
#define SK0 224           // 7*32  : padded 3*66
#define SK1 384           // 12*32 : 3*128

#define HAVE_TDM (__has_builtin(__builtin_amdgcn_tensor_load_to_lds))

union FragU { f32x4 f4[2]; v16h h; };
union AccU  { v8f v; float f[8]; };

// -------------------------- gfx1250 helpers --------------------------------

// Fallback async copy (verified to assemble on this toolchain in round 1).
__device__ __forceinline__ void cdna5_async_b128_to_lds(const _Float16* g, _Float16* l) {
#if defined(__HIP_DEVICE_COMPILE__)
  unsigned lds_off = (unsigned)(uintptr_t)(void*)l;
  asm volatile("global_load_async_to_lds_b128 %0, %1, off"
               :: "v"(lds_off), "v"(g) : "memory");
#else
  (void)g; (void)l;
#endif
}

__device__ __forceinline__ void cdna5_wait_async0() {
#if defined(__HIP_DEVICE_COMPILE__)
  asm volatile("s_wait_asynccnt 0" ::: "memory");
#endif
}

// TDM: load a 2D f16 tile [NH rows x skp cols], row stride skp, into LDS.
// D# layout per CDNA5 ISA (count=1, type=2,
// data_size=1 (2 bytes), tensor_dim0=tile_dim0=skp, tensor_dim1=tile_dim1=NH).
// This toolchain exposes the 6-arg builtin:
//   (uint32x4 g0, int32x8 g1, int32x4 g2, int32x4 g3, int32x8 g4, i32 cpol)
__device__ __forceinline__ void tdm_load_b_to_lds(const _Float16* gsrc,
                                                  unsigned lds_off, int skp) {
#if defined(__HIP_DEVICE_COMPILE__) && HAVE_TDM
  unsigned long long ga = (unsigned long long)(uintptr_t)gsrc;
  u32x4 g0 = { 1u,                                   // count=1 (valid D#)
               lds_off,                               // lds_addr
               (unsigned)ga,                          // global_addr[31:0]
               (unsigned)((ga >> 32) & 0x1FFFFFFu)    // global_addr[56:32]
                 | 0x80000000u };                     // type=2 ("image")
  i32x8 g1 = { 0x10000,                              // data_size=1 (2B), mask=0
               (int)((skp & 0xFFFF) << 16),          // tensor_dim0[15:0] @bits[63:48]
               (NH << 16),                           // tensor_dim1[15:0] @bits[95:80]
               (int)(skp << 16),                     // tile_dim0 @bits[127:112]
               NH,                                   // tile_dim1 @bits[143:128]
               skp,                                  // tensor_dim0_stride[31:0]
               0, 0 };
  i32x4 gz4 = { 0, 0, 0, 0 };
  i32x8 gz8 = { 0, 0, 0, 0, 0, 0, 0, 0 };
  __builtin_amdgcn_tensor_load_to_lds(g0, g1, gz4, gz4, gz8, 0);
#else
  (void)gsrc; (void)lds_off; (void)skp;
#endif
}

__device__ __forceinline__ void tdm_wait0() {
#if defined(__HIP_DEVICE_COMPILE__)
#if __has_builtin(__builtin_amdgcn_s_wait_tensorcnt)
  __builtin_amdgcn_s_wait_tensorcnt(0);
#else
  asm volatile("s_wait_tensorcnt 0" ::: "memory");
#endif
#endif
}

__device__ __forceinline__ v16h load_frag16(const _Float16* p) {
  FragU u;
  u.f4[0] = ((const f32x4*)p)[0];   // halves [0..7]
  u.f4[1] = ((const f32x4*)p)[2];   // halves [16..23]
  return u.h;
}

// ----------------------------- graph prep ----------------------------------

__global__ void deg_kernel(const int* __restrict__ dst, const float* __restrict__ w,
                           float* __restrict__ deg) {
  int e = blockIdx.x * blockDim.x + threadIdx.x;
  if (e < NE) atomicAdd(&deg[dst[e]], w[e]);
}

__global__ void norm_kernel(const int* __restrict__ dst, const float* __restrict__ w,
                            const float* __restrict__ deg, float* __restrict__ ewn) {
  int e = blockIdx.x * blockDim.x + threadIdx.x;
  if (e < NE) ewn[e] = w[e] / fmaxf(deg[dst[e]], 1e-6f);
}

// W [K,Fin,NH] fp32 -> Bt [NH, SKP] f16 (B^T, K folded, zero padded)
__global__ void wcvt_kernel(const float* __restrict__ W, _Float16* __restrict__ Bt,
                            int Fin, int SKP) {
  int tid = blockIdx.x * blockDim.x + threadIdx.x;
  if (tid >= NH * SKP) return;
  int n = tid / SKP, c = tid % SKP;
  float v = 0.f;
  if (c < 3 * Fin) {
    int k = c / Fin, f = c % Fin;
    v = W[((size_t)k * Fin + f) * NH + n];
  }
  Bt[(size_t)n * SKP + c] = (_Float16)v;
}

// ----------------------------- per-step ops --------------------------------

__global__ void concat_kernel(const float* __restrict__ xp, const float* __restrict__ hp,
                              float* __restrict__ T0, _Float16* __restrict__ A,
                              int Fin, int Fx, int SK, int isSeq, int t) {
  int tid = blockIdx.x * blockDim.x + threadIdx.x;
  if (tid >= BN * Fin) return;
  int row = tid / Fin, f = tid % Fin;
  float v;
  if (f < Fx) {
    if (isSeq) v = xp[((size_t)row * NT_STEPS + t) * NF + f];   // x_seq [B,N,T,F]
    else       v = xp[(size_t)row * NH + f];                    // previous layer h
  } else {
    v = hp[(size_t)row * NH + (f - Fx)];
  }
  T0[(size_t)row * Fin + f] = v;
  A[(size_t)row * SK + f] = (_Float16)v;
}

__global__ void spmm_kernel(const float* __restrict__ Tp, float* __restrict__ Tn,
                            const int* __restrict__ src, const int* __restrict__ dst,
                            const float* __restrict__ ewn, int Fin) {
  long tid = (long)blockIdx.x * blockDim.x + threadIdx.x;
  if (tid >= (long)NE * Fin) return;
  int e = (int)(tid / Fin), f = (int)(tid % Fin);
  int s = src[e], d = dst[e];
  float w = ewn[e];
  const float* ps = Tp + (size_t)s * Fin + f;
  float* pd = Tn + (size_t)d * Fin + f;
#pragma unroll
  for (int b = 0; b < NB; ++b)
    atomicAdd(pd + (size_t)b * NN * Fin, w * ps[(size_t)b * NN * Fin]);
}

__global__ void cvt_kernel(const float* __restrict__ Tk, _Float16* __restrict__ A,
                           int Fin, int SK, int slot) {
  int tid = blockIdx.x * blockDim.x + threadIdx.x;
  if (tid >= BN * Fin) return;
  int row = tid / Fin, f = tid % Fin;
  A[(size_t)row * SK + slot * Fin + f] = (_Float16)Tk[tid];
}

// ------------------------------ WMMA GEMMs ---------------------------------
// Block: 256 threads = 8 waves; wave owns 16 rows x 64 cols (4 N-tiles).
// grid = BN/128 = 625.  Whole B^T resides in LDS (TDM), K-loop barrier-free.

template <int SK, int NG>
__device__ __forceinline__ void stage_b_lds(_Float16* ldsB, const _Float16* bt0,
                                            const _Float16* bt1, int tid, int wave) {
  constexpr int SKP = SK + 8;
#if defined(__HIP_DEVICE_COMPILE__) && HAVE_TDM
  if (wave == 0) {
    tdm_load_b_to_lds(bt0, (unsigned)(uintptr_t)(void*)ldsB, SKP);
    if (NG == 2)
      tdm_load_b_to_lds(bt1, (unsigned)(uintptr_t)(void*)(ldsB + (size_t)NH * SKP), SKP);
    tdm_wait0();
  }
  (void)tid;
#else
  constexpr int CHUNKS = NG * NH * SKP / 8;     // 8-half (16B) chunks
  for (int c = tid; c < CHUNKS; c += 256) {
    int g = c / (NH * SKP / 8);
    int off = (c - g * (NH * SKP / 8)) * 8;
    const _Float16* gp = (g ? bt1 : bt0) + off;
    cdna5_async_b128_to_lds(gp, ldsB + (size_t)g * NH * SKP + off);
  }
  cdna5_wait_async0();
  (void)wave;
#endif
  __syncthreads();
}

template <int SK>
__global__ void __launch_bounds__(256) gemm_zr_kernel(
    const _Float16* A, const _Float16* __restrict__ Btz, const _Float16* __restrict__ Btr,
    const float* __restrict__ bz, const float* __restrict__ br,
    const float* __restrict__ h, float* __restrict__ zb,
    float* __restrict__ T0, _Float16* Aout, int Fin, int Fx)
{
  constexpr int SKP = SK + 8;
  extern __shared__ char smem[];
  _Float16* ldsB = (_Float16*)smem;             // [2][NH][SKP]

  const int tid   = threadIdx.x;
  const int wave  = tid >> 5;
  const int lane  = tid & 31;
  const int halfq = lane >> 4;
  const int r     = lane & 15;
  const int rowBase = (blockIdx.x * 8 + wave) * 16;

  stage_b_lds<SK, 2>(ldsB, Btz, Btr, tid, wave);

  v8f accZ[4] = {};
  v8f accR[4] = {};

#pragma unroll
  for (int kb = 0; kb < SK; kb += 32) {
    v16h bzf[4], brf[4];
#pragma unroll
    for (int nt = 0; nt < 4; ++nt) {
      const int n = nt * 16 + r;
      bzf[nt] = load_frag16(&ldsB[(size_t)n * SKP + kb + halfq * 8]);
      brf[nt] = load_frag16(&ldsB[(size_t)(NH + n) * SKP + kb + halfq * 8]);
    }
    v16h afrag = load_frag16(A + (size_t)(rowBase + r) * SK + kb + halfq * 8);
#if defined(__HIP_DEVICE_COMPILE__)
#pragma unroll
    for (int nt = 0; nt < 4; ++nt)
      accZ[nt] = __builtin_amdgcn_wmma_f32_16x16x32_f16(false, afrag, false, bzf[nt],
                                                        (short)0, accZ[nt], false, false);
#pragma unroll
    for (int nt = 0; nt < 4; ++nt)
      accR[nt] = __builtin_amdgcn_wmma_f32_16x16x32_f16(false, afrag, false, brf[nt],
                                                        (short)0, accR[nt], false, false);
#endif
  }

  // GRU z/r epilogue: z=sigmoid(.+bz), r=sigmoid(.+br); write z, and r*h into
  // fp32 candidate-chain input T0 and f16 A slot 0.
#pragma unroll
  for (int nt = 0; nt < 4; ++nt) {
    AccU uz, ur; uz.v = accZ[nt]; ur.v = accR[nt];
    const int col = nt * 16 + r;
    const float bzv = bz[col], brv = br[col];
#pragma unroll
    for (int v = 0; v < 8; ++v) {
      const int row = rowBase + halfq * 8 + v;
      const size_t idx = (size_t)row * NH + col;
      float zv = 1.f / (1.f + expf(-(uz.f[v] + bzv)));
      float rv = 1.f / (1.f + expf(-(ur.f[v] + brv)));
      zb[idx] = zv;
      float rh = rv * h[idx];
      T0[(size_t)row * Fin + Fx + col] = rh;
      Aout[(size_t)row * SK + Fx + col] = (_Float16)rh;
    }
  }
}

template <int SK>
__global__ void __launch_bounds__(256) gemm_c_kernel(
    const _Float16* __restrict__ A, const _Float16* __restrict__ Btc,
    const float* __restrict__ bc, const float* __restrict__ zb,
    float* h)
{
  constexpr int SKP = SK + 8;
  extern __shared__ char smem[];
  _Float16* ldsB = (_Float16*)smem;             // [NH][SKP]

  const int tid   = threadIdx.x;
  const int wave  = tid >> 5;
  const int lane  = tid & 31;
  const int halfq = lane >> 4;
  const int r     = lane & 15;
  const int rowBase = (blockIdx.x * 8 + wave) * 16;

  stage_b_lds<SK, 1>(ldsB, Btc, Btc, tid, wave);

  v8f acc[4] = {};

#pragma unroll
  for (int kb = 0; kb < SK; kb += 32) {
    v16h bf[4];
#pragma unroll
    for (int nt = 0; nt < 4; ++nt)
      bf[nt] = load_frag16(&ldsB[(size_t)(nt * 16 + r) * SKP + kb + halfq * 8]);
    v16h afrag = load_frag16(A + (size_t)(rowBase + r) * SK + kb + halfq * 8);
#if defined(__HIP_DEVICE_COMPILE__)
#pragma unroll
    for (int nt = 0; nt < 4; ++nt)
      acc[nt] = __builtin_amdgcn_wmma_f32_16x16x32_f16(false, afrag, false, bf[nt],
                                                       (short)0, acc[nt], false, false);
#endif
  }

  // GRU h update: c = tanh(.+bc); h = z*h + (1-z)*c
#pragma unroll
  for (int nt = 0; nt < 4; ++nt) {
    AccU uc; uc.v = acc[nt];
    const int col = nt * 16 + r;
    const float bcv = bc[col];
#pragma unroll
    for (int v = 0; v < 8; ++v) {
      const int row = rowBase + halfq * 8 + v;
      const size_t idx = (size_t)row * NH + col;
      float cv = tanhf(uc.f[v] + bcv);
      float zv = zb[idx];
      h[idx] = zv * h[idx] + (1.f - zv) * cv;
    }
  }
}

// ------------------------------- launcher ----------------------------------

extern "C" void kernel_launch(void* const* d_in, const int* in_sizes, int n_in,
                              void* d_out, int out_size, void* d_ws, size_t ws_size,
                              hipStream_t stream) {
  (void)in_sizes; (void)n_in; (void)out_size; (void)ws_size;

  const float* x_seq  = (const float*)d_in[0];
  const int*   ei     = (const int*)d_in[1];
  const float* ew_raw = (const float*)d_in[2];
  const float* Wp[2][3]   = {{(const float*)d_in[3],  (const float*)d_in[5],  (const float*)d_in[7]},
                             {(const float*)d_in[9],  (const float*)d_in[11], (const float*)d_in[13]}};
  const float* biasP[2][3] = {{(const float*)d_in[4],  (const float*)d_in[6],  (const float*)d_in[8]},
                              {(const float*)d_in[10], (const float*)d_in[12], (const float*)d_in[14]}};
  const int* srcI = ei;
  const int* dstI = ei + NE;

  char* ws = (char*)d_ws;
  size_t cur = 0;
  auto alloc = [&](size_t bytes) -> void* {
    void* p = ws + cur;
    cur = (cur + bytes + 255) & ~(size_t)255;
    return p;
  };

  const int FinL[2] = {FIN0, FIN1};
  const int FxL[2]  = {NF, NH};
  const int SKL[2]  = {SK0, SK1};
  const int SKPL[2] = {SK0 + 8, SK1 + 8};

  float*    ewn  = (float*)alloc((size_t)NE * 4);
  float*    deg  = (float*)alloc((size_t)NN * 4);
  float*    h0   = (float*)alloc((size_t)BN * NH * 4);
  float*    h1   = (float*)alloc((size_t)BN * NH * 4);
  float*    zbuf = (float*)alloc((size_t)BN * NH * 4);
  float*    T0   = (float*)alloc((size_t)BN * FIN1 * 4);
  float*    T1   = (float*)alloc((size_t)BN * FIN1 * 4);
  float*    T2   = (float*)alloc((size_t)BN * FIN1 * 4);
  _Float16* A0   = (_Float16*)alloc((size_t)BN * SK0 * 2);
  _Float16* A1   = (_Float16*)alloc((size_t)BN * SK1 * 2);
  _Float16* Bt[2][3];
  for (int l = 0; l < 2; ++l)
    for (int g = 0; g < 3; ++g)
      Bt[l][g] = (_Float16*)alloc((size_t)NH * SKPL[l] * 2);

  _Float16* AL[2] = {A0, A1};
  float*    hL[2] = {h0, h1};

  // dynamic-LDS sizes (layer1 z/r kernel needs ~98KB; gfx1250 WGP has 320KB)
  const size_t ldsZR[2] = {(size_t)2 * NH * SKPL[0] * 2, (size_t)2 * NH * SKPL[1] * 2};
  const size_t ldsC[2]  = {(size_t)NH * SKPL[0] * 2,     (size_t)NH * SKPL[1] * 2};
  (void)hipFuncSetAttribute((const void*)gemm_zr_kernel<SK0>, hipFuncAttributeMaxDynamicSharedMemorySize, (int)ldsZR[0]);
  (void)hipFuncSetAttribute((const void*)gemm_zr_kernel<SK1>, hipFuncAttributeMaxDynamicSharedMemorySize, (int)ldsZR[1]);
  (void)hipFuncSetAttribute((const void*)gemm_c_kernel<SK0>,  hipFuncAttributeMaxDynamicSharedMemorySize, (int)ldsC[0]);
  (void)hipFuncSetAttribute((const void*)gemm_c_kernel<SK1>,  hipFuncAttributeMaxDynamicSharedMemorySize, (int)ldsC[1]);

  // --- graph normalization (random-walk) ---
  (void)hipMemsetAsync(deg, 0, (size_t)NN * 4, stream);
  deg_kernel<<<(NE + 255) / 256, 256, 0, stream>>>(dstI, ew_raw, deg);
  norm_kernel<<<(NE + 255) / 256, 256, 0, stream>>>(dstI, ew_raw, deg, ewn);

  // --- weight transpose + f16 convert (padded stride SK+8) ---
  for (int l = 0; l < 2; ++l)
    for (int g = 0; g < 3; ++g)
      wcvt_kernel<<<(NH * SKPL[l] + 255) / 256, 256, 0, stream>>>(Wp[l][g], Bt[l][g],
                                                                  FinL[l], SKPL[l]);

  // --- init state; zero A buffers so pad columns stay zero all call ---
  (void)hipMemsetAsync(h0, 0, (size_t)BN * NH * 4, stream);
  (void)hipMemsetAsync(h1, 0, (size_t)BN * NH * 4, stream);
  (void)hipMemsetAsync(A0, 0, (size_t)BN * SK0 * 2, stream);
  (void)hipMemsetAsync(A1, 0, (size_t)BN * SK1 * 2, stream);

  for (int t = 0; t < NT_STEPS; ++t) {
    for (int l = 0; l < 2; ++l) {
      const int Fin = FinL[l], Fx = FxL[l], SK = SKL[l];
      _Float16* A = AL[l];
      float* h = hL[l];
      const float* xp = (l == 0) ? x_seq : h0;
      const int catBlocks = (BN * Fin + 255) / 256;
      const int spBlocks  = (int)(((long)NE * Fin + 255) / 256);
      const size_t Tbytes = (size_t)BN * Fin * 4;

      concat_kernel<<<catBlocks, 256, 0, stream>>>(xp, h, T0, A, Fin, Fx, SK,
                                                   (l == 0) ? 1 : 0, t);
      // z/r diffusion chain: slots 1,2
      (void)hipMemsetAsync(T1, 0, Tbytes, stream);
      spmm_kernel<<<spBlocks, 256, 0, stream>>>(T0, T1, srcI, dstI, ewn, Fin);
      cvt_kernel<<<catBlocks, 256, 0, stream>>>(T1, A, Fin, SK, 1);
      (void)hipMemsetAsync(T2, 0, Tbytes, stream);
      spmm_kernel<<<spBlocks, 256, 0, stream>>>(T1, T2, srcI, dstI, ewn, Fin);
      cvt_kernel<<<catBlocks, 256, 0, stream>>>(T2, A, Fin, SK, 2);

      // fused z & r GEMM + epilogue
      if (l == 0)
        gemm_zr_kernel<SK0><<<BN / 128, 256, ldsZR[0], stream>>>(A, Bt[l][0], Bt[l][1],
            biasP[l][0], biasP[l][1], h, zbuf, T0, A, Fin, Fx);
      else
        gemm_zr_kernel<SK1><<<BN / 128, 256, ldsZR[1], stream>>>(A, Bt[l][0], Bt[l][1],
            biasP[l][0], biasP[l][1], h, zbuf, T0, A, Fin, Fx);

      // candidate diffusion chain on [x, r*h]
      (void)hipMemsetAsync(T1, 0, Tbytes, stream);
      spmm_kernel<<<spBlocks, 256, 0, stream>>>(T0, T1, srcI, dstI, ewn, Fin);
      cvt_kernel<<<catBlocks, 256, 0, stream>>>(T1, A, Fin, SK, 1);
      (void)hipMemsetAsync(T2, 0, Tbytes, stream);
      spmm_kernel<<<spBlocks, 256, 0, stream>>>(T1, T2, srcI, dstI, ewn, Fin);
      cvt_kernel<<<catBlocks, 256, 0, stream>>>(T2, A, Fin, SK, 2);

      // c GEMM + GRU state update (in place on h)
      if (l == 0)
        gemm_c_kernel<SK0><<<BN / 128, 256, ldsC[0], stream>>>(A, Bt[l][2], biasP[l][2], zbuf, h);
      else
        gemm_c_kernel<SK1><<<BN / 128, 256, ldsC[1], stream>>>(A, Bt[l][2], biasP[l][2], zbuf, h);
    }
  }

  // output: [2, B, N, H] = [h0_final ; h1_final]
  (void)hipMemcpyAsync(d_out, h0, (size_t)BN * NH * 4, hipMemcpyDeviceToDevice, stream);
  (void)hipMemcpyAsync((float*)d_out + (size_t)BN * NH, h1, (size_t)BN * NH * 4,
                       hipMemcpyDeviceToDevice, stream);
}
